// CausalSelfAttention_7464653161057
// MI455X (gfx1250) — compile-verified, matrix operand order checked by reading, not run on hardware
//
#include <hip/hip_runtime.h>

typedef __attribute__((ext_vector_type(16))) __bf16 v16bf;
typedef __attribute__((ext_vector_type(8)))  float  v8f;
typedef unsigned short u16;
typedef unsigned int   u32;

#define D_MODEL 1024
#define N_HEADS 16
#define D_HEAD  64
#define BATCH   4
#define SEQ     2048
#define MTOT    (BATCH*SEQ)   // 8192 tokens

__device__ __forceinline__ u16 f2bf(float f) {
  u32 u = __float_as_uint(f);
  u += 0x7fffu + ((u >> 16) & 1u);   // round-to-nearest-even
  return (u16)(u >> 16);
}

// ---------------------------------------------------------------- convert
__global__ void cvt_bf16_kernel(const float* __restrict__ src,
                                u16* __restrict__ dst, int n) {
  int i = blockIdx.x * blockDim.x + threadIdx.x;
  if (i < n) dst[i] = f2bf(src[i]);
}

// ---------------------------------------------------------------- fused QKV GEMM
// out[m][n] = sum_k xb[m][k] * W[n][k], N = 3*1024 (Q|K|V).
// Each wave: 32(M) x 64(N) tile, K=1024 in steps of 32 -> 256 WMMAs.
__global__ __launch_bounds__(256)
void qkv_gemm_kernel(const u16* __restrict__ xb,
                     const u16* __restrict__ wq,
                     const u16* __restrict__ wk,
                     const u16* __restrict__ wv,
                     u16* __restrict__ Qh,   // [b][h][t][64]
                     u16* __restrict__ Kh,   // [b][h][t][64]
                     u16* __restrict__ Vt) { // [b][h][64][t]
  const int lane = threadIdx.x & 31;
  const int wave = blockIdx.x * (blockDim.x >> 5) + (threadIdx.x >> 5);
  const int tilesN = (3 * D_MODEL) / 64;          // 48
  const int tm = (wave / tilesN) * 32;
  const int tn = (wave % tilesN) * 64;
  const int llo = lane & 15, lhi = lane >> 4;

  const u16* wsrc = (tn < D_MODEL) ? wq : (tn < 2*D_MODEL ? wk : wv);
  const int  nbase = tn & (D_MODEL - 1);

  v8f acc[2][4] = {};
  const u16* arow0 = xb + (tm + llo) * D_MODEL + lhi * 16;
  const u16* arow1 = arow0 + 16 * D_MODEL;
  const u16* brow[4];
#pragma unroll
  for (int j = 0; j < 4; j++)
    brow[j] = wsrc + (nbase + j * 16 + llo) * D_MODEL + lhi * 16;

  for (int k0 = 0; k0 < D_MODEL; k0 += 32) {
    v16bf a0 = *(const v16bf*)(arow0 + k0);
    v16bf a1 = *(const v16bf*)(arow1 + k0);
#pragma unroll
    for (int j = 0; j < 4; j++) {
      v16bf b = *(const v16bf*)(brow[j] + k0);
      acc[0][j] = __builtin_amdgcn_wmma_f32_16x16x32_bf16(
          false, a0, false, b, (short)0, acc[0][j], false, false);
      acc[1][j] = __builtin_amdgcn_wmma_f32_16x16x32_bf16(
          false, a1, false, b, (short)0, acc[1][j], false, false);
    }
  }

#pragma unroll
  for (int i = 0; i < 2; i++)
#pragma unroll
    for (int j = 0; j < 4; j++)
#pragma unroll
      for (int e = 0; e < 8; e++) {
        int m = tm + i * 16 + e + (lhi << 3);
        int n = tn + j * 16 + llo;
        int b = m >> 11;           // /SEQ
        int t = m & (SEQ - 1);
        u16 v = f2bf(acc[i][j][e]);
        if (n < D_MODEL) {
          int h = n >> 6, d = n & 63;
          Qh[((b * N_HEADS + h) * SEQ + t) * D_HEAD + d] = v;
        } else if (n < 2 * D_MODEL) {
          int c = n - D_MODEL, h = c >> 6, d = c & 63;
          Kh[((b * N_HEADS + h) * SEQ + t) * D_HEAD + d] = v;
        } else {
          int c = n - 2 * D_MODEL, h = c >> 6, d = c & 63;
          Vt[((b * N_HEADS + h) * D_HEAD + d) * SEQ + t] = v;   // transposed
        }
      }
}

// ---------------------------------------------------------------- flash attention
// One wave per (head-instance, 16-query tile); online softmax over 32-key blocks.
__global__ __launch_bounds__(256)
void attn_kernel(const u16* __restrict__ Qh, const u16* __restrict__ Kh,
                 const u16* __restrict__ Vt, u16* __restrict__ Ob) {
  __shared__ u16 pbuf[8][16 * 32];                 // per-wave P transpose buffer
  const int lane = threadIdx.x & 31;
  const int wv_  = threadIdx.x >> 5;
  const int wave = blockIdx.x * 8 + wv_;
  const int qt = wave & 127;                       // query tile (T/16)
  const int bh = wave >> 7;                        // b*16 + h
  const int llo = lane & 15, lhi = lane >> 4;
  const float SCALE = 0.125f;                      // 1/sqrt(64)

  const u16* qbase = Qh + (bh * SEQ + qt * 16) * D_HEAD;
  v16bf qf[2];
#pragma unroll
  for (int ds = 0; ds < 2; ds++)
    qf[ds] = *(const v16bf*)(qbase + llo * D_HEAD + ds * 32 + lhi * 16);

  float mrow[8], lrow[8];
  v8f o[4] = {};
#pragma unroll
  for (int e = 0; e < 8; e++) { mrow[e] = -1e30f; lrow[e] = 0.f; }

  const int nblk = (qt + 2) >> 1;                  // ceil(16*(qt+1)/32)
  for (int kb = 0; kb < nblk; kb++) {
    const int k0 = kb * 32;
    v8f s0 = {}, s1 = {};
#pragma unroll
    for (int ds = 0; ds < 2; ds++) {
      v16bf kf0 = *(const v16bf*)(Kh + (bh * SEQ + k0 + llo) * D_HEAD + ds * 32 + lhi * 16);
      v16bf kf1 = *(const v16bf*)(Kh + (bh * SEQ + k0 + 16 + llo) * D_HEAD + ds * 32 + lhi * 16);
      s0 = __builtin_amdgcn_wmma_f32_16x16x32_bf16(false, qf[ds], false, kf0, (short)0, s0, false, false);
      s1 = __builtin_amdgcn_wmma_f32_16x16x32_bf16(false, qf[ds], false, kf1, (short)0, s1, false, false);
    }
    const bool lastb = (kb == nblk - 1);
    float p0[8], p1[8];
#pragma unroll
    for (int e = 0; e < 8; e++) {
      int qrow = qt * 16 + e + (lhi << 3);
      float v0 = s0[e] * SCALE, v1 = s1[e] * SCALE;
      if (lastb) {                                 // causal mask, diagonal block only
        if (k0 + llo      > qrow) v0 = -3.0e38f;
        if (k0 + 16 + llo > qrow) v1 = -3.0e38f;
      }
      p0[e] = v0; p1[e] = v1;
    }
    // online softmax: row reductions across the 16-lane half holding this row
#pragma unroll
    for (int e = 0; e < 8; e++) {
      float mx = fmaxf(p0[e], p1[e]);
#pragma unroll
      for (int off = 1; off < 16; off <<= 1) mx = fmaxf(mx, __shfl_xor(mx, off, 32));
      float mnew = fmaxf(mrow[e], mx);
      float corr = __expf(mrow[e] - mnew);
      float e0 = __expf(p0[e] - mnew);
      float e1 = __expf(p1[e] - mnew);
      float ls = e0 + e1;
#pragma unroll
      for (int off = 1; off < 16; off <<= 1) ls += __shfl_xor(ls, off, 32);
      lrow[e] = lrow[e] * corr + ls;
      mrow[e] = mnew;
#pragma unroll
      for (int j = 0; j < 4; j++) o[j][e] *= corr;
      p0[e] = e0; p1[e] = e1;
    }
    // P (C-layout) -> LDS bf16 -> A-fragment layout
    u16* pb = pbuf[wv_];
#pragma unroll
    for (int e = 0; e < 8; e++) {
      int r = e + (lhi << 3);
      pb[r * 32 + llo]      = f2bf(p0[e]);
      pb[r * 32 + 16 + llo] = f2bf(p1[e]);
    }
    asm volatile("s_wait_dscnt 0x0" ::: "memory");
    v16bf pf = *(const v16bf*)(pb + llo * 32 + lhi * 16);
#pragma unroll
    for (int j = 0; j < 4; j++) {
      v16bf vf = *(const v16bf*)(Vt + (bh * D_HEAD + j * 16 + llo) * SEQ + k0 + lhi * 16);
      o[j] = __builtin_amdgcn_wmma_f32_16x16x32_bf16(false, pf, false, vf, (short)0, o[j], false, false);
    }
  }

  const int b = bh >> 4, h = bh & 15;
#pragma unroll
  for (int e = 0; e < 8; e++) {
    float inv = 1.0f / lrow[e];
    int t = qt * 16 + e + (lhi << 3);
#pragma unroll
    for (int j = 0; j < 4; j++) {
      int c = h * D_HEAD + j * 16 + llo;
      Ob[(b * SEQ + t) * D_MODEL + c] = f2bf(o[j][e] * inv);
    }
  }
}

// ---------------------------------------------------------------- output projection
__global__ __launch_bounds__(256)
void out_gemm_kernel(const u16* __restrict__ Ob, const u16* __restrict__ wo,
                     float* __restrict__ out) {
  const int lane = threadIdx.x & 31;
  const int wave = blockIdx.x * (blockDim.x >> 5) + (threadIdx.x >> 5);
  const int tilesN = D_MODEL / 64;                 // 16
  const int tm = (wave / tilesN) * 32;
  const int tn = (wave % tilesN) * 64;
  const int llo = lane & 15, lhi = lane >> 4;

  v8f acc[2][4] = {};
  const u16* arow0 = Ob + (tm + llo) * D_MODEL + lhi * 16;
  const u16* arow1 = arow0 + 16 * D_MODEL;
  const u16* brow[4];
#pragma unroll
  for (int j = 0; j < 4; j++)
    brow[j] = wo + (tn + j * 16 + llo) * D_MODEL + lhi * 16;

  for (int k0 = 0; k0 < D_MODEL; k0 += 32) {
    v16bf a0 = *(const v16bf*)(arow0 + k0);
    v16bf a1 = *(const v16bf*)(arow1 + k0);
#pragma unroll
    for (int j = 0; j < 4; j++) {
      v16bf b = *(const v16bf*)(brow[j] + k0);
      acc[0][j] = __builtin_amdgcn_wmma_f32_16x16x32_bf16(
          false, a0, false, b, (short)0, acc[0][j], false, false);
      acc[1][j] = __builtin_amdgcn_wmma_f32_16x16x32_bf16(
          false, a1, false, b, (short)0, acc[1][j], false, false);
    }
  }
#pragma unroll
  for (int i = 0; i < 2; i++)
#pragma unroll
    for (int j = 0; j < 4; j++)
#pragma unroll
      for (int e = 0; e < 8; e++) {
        int m = tm + i * 16 + e + (lhi << 3);
        int n = tn + j * 16 + llo;
        out[m * D_MODEL + n] = acc[i][j][e];
      }
}

// ---------------------------------------------------------------- launch
extern "C" void kernel_launch(void* const* d_in, const int* in_sizes, int n_in,
                              void* d_out, int out_size, void* d_ws, size_t ws_size,
                              hipStream_t stream) {
  (void)in_sizes; (void)n_in; (void)out_size; (void)ws_size;
  const float* x  = (const float*)d_in[0];
  const float* Wq = (const float*)d_in[1];
  const float* Wk = (const float*)d_in[2];
  const float* Wv = (const float*)d_in[3];
  const float* Wo = (const float*)d_in[4];
  float* out = (float*)d_out;

  const size_t MB = 1024 * 1024;
  char* ws = (char*)d_ws;
  u16* xb  = (u16*)(ws);             // 16 MB  x in bf16
  u16* wqb = (u16*)(ws + 16 * MB);   //  2 MB
  u16* wkb = (u16*)(ws + 18 * MB);   //  2 MB
  u16* wvb = (u16*)(ws + 20 * MB);   //  2 MB
  u16* wob = (u16*)(ws + 22 * MB);   //  2 MB
  u16* Qh  = (u16*)(ws + 24 * MB);   // 16 MB  [b][h][t][64]
  u16* Kh  = (u16*)(ws + 40 * MB);   // 16 MB  [b][h][t][64]
  u16* Vt  = (u16*)(ws + 56 * MB);   // 16 MB  [b][h][64][t]
  u16* Ob  = (u16*)(ws + 72 * MB);   // 16 MB  [b][t][1024]  (ends at 88 MB)

  const int nx = MTOT * D_MODEL;     // 8,388,608
  const int nw = D_MODEL * D_MODEL;  // 1,048,576
  cvt_bf16_kernel<<<nx / 256, 256, 0, stream>>>(x,  xb,  nx);
  cvt_bf16_kernel<<<nw / 256, 256, 0, stream>>>(Wq, wqb, nw);
  cvt_bf16_kernel<<<nw / 256, 256, 0, stream>>>(Wk, wkb, nw);
  cvt_bf16_kernel<<<nw / 256, 256, 0, stream>>>(Wv, wvb, nw);
  cvt_bf16_kernel<<<nw / 256, 256, 0, stream>>>(Wo, wob, nw);

  qkv_gemm_kernel<<<(256 * 48) / 8, 256, 0, stream>>>(xb, wqb, wkb, wvb, Qh, Kh, Vt);
  attn_kernel    <<<(64 * 128) / 8, 256, 0, stream>>>(Qh, Kh, Vt, Ob);
  out_gemm_kernel<<<(256 * 16) / 8, 256, 0, stream>>>(Ob, wob, out);
}
